// MLP_27058293964900
// MI455X (gfx1250) — matrix operand order repaired and checked by reading, not compiled, not run on hardware
//
#include <hip/hip_runtime.h>
#include <hip/hip_bf16.h>
#include <math.h>
#include <stdint.h>

// ---------------- CDNA5 WMMA types ----------------
typedef __attribute__((ext_vector_type(16))) __bf16 v16bf;
typedef __attribute__((ext_vector_type(8)))  float  v8f;
typedef __attribute__((ext_vector_type(4)))  unsigned int u32x4;

union Frag {
  v16bf v;
  unsigned short us[16];
  u32x4 q[2];
};

__device__ __forceinline__ unsigned short f2bf(float f) {
  unsigned u = __float_as_uint(f);
  unsigned r = (u + 0x7FFFu + ((u >> 16) & 1u)) >> 16;   // round-to-nearest-even
  return (unsigned short)r;
}
__device__ __forceinline__ float bf2f(unsigned short h) {
  return __uint_as_float(((unsigned)h) << 16);
}

// ---- CDNA5 async global->LDS copy (ASYNCcnt-tracked), per CDNA5 ISA ----
__device__ __forceinline__ void async_ld_b128(unsigned lds_off, const void* gptr) {
  asm volatile("global_load_async_to_lds_b128 %0, %1, off"
               :: "v"(lds_off), "v"((unsigned long long)(uintptr_t)gptr)
               : "memory");
}
__device__ __forceinline__ void wait_async0() {
#if __has_builtin(__builtin_amdgcn_s_wait_asynccnt)
  __builtin_amdgcn_s_wait_asynccnt(0);
#else
  asm volatile("s_wait_asynccnt 0x0" ::: "memory");
#endif
}

#define EPSV 1e-5f
#define N_PAIR 5000
#define N_IMG  100
#define PPI    50

// ---------------- A fetchers: prep(row) once, then load8(k) per K-step ----------------
struct ADirect {           // A bf16 [M][K] row-major
  const unsigned short* A; int K; const unsigned short* base;
  __device__ __forceinline__ void prep(int row) {
    base = A + (size_t)row * K;
    __builtin_prefetch(base, 0, 0);                // global_prefetch: warm the streamed row
  }
  __device__ __forceinline__ void load8(int k, unsigned short* d) const {
    *(u32x4*)d = *(const u32x4*)(base + k);
  }
};
struct AGather {           // A = feat_bf16[pair_idx[row][which]]  (K=2048)
  const unsigned short* feat; const int* pair; int which; const unsigned short* base;
  __device__ __forceinline__ void prep(int row) {
    base = feat + (size_t)pair[row * 2 + which] * 2048;
    __builtin_prefetch(base, 0, 0);
  }
  __device__ __forceinline__ void load8(int k, unsigned short* d) const {
    *(u32x4*)d = *(const u32x4*)(base + k);
  }
};
struct AUnion {            // row=(p,hw), k=c; union_feat f32 [p][1024][49]
  const float* uf; const float* base;
  __device__ __forceinline__ void prep(int row) {
    int p = row / 49, hw = row % 49;
    base = uf + (size_t)p * 1024 * 49 + hw;
  }
  __device__ __forceinline__ void load8(int k, unsigned short* d) const {
    const float* b = base + (size_t)k * 49;
#pragma unroll
    for (int i = 0; i < 8; ++i) d[i] = f2bf(b[i * 49]);
  }
};
struct AConv1 {            // im2col of spatial_masks [p][2][27][27]; k = c*49 + ky*7+kx (K pad 128)
  const float* sm; const float* smp; int iy0, ix0;
  __device__ __forceinline__ void prep(int row) {
    int p = row / 196, r = row % 196, oy = r / 14, ox = r % 14;
    smp = sm + (size_t)p * 2 * 729;
    iy0 = oy * 2 - 3; ix0 = ox * 2 - 3;
  }
  __device__ __forceinline__ void load8(int k, unsigned short* d) const {
#pragma unroll
    for (int i = 0; i < 8; ++i) {
      int kk = k + i; float v = 0.f;
      if (kk < 98) {
        int c = kk / 49, t = kk % 49, ky = t / 7, kx = t % 7;
        int iy = iy0 + ky, ix = ix0 + kx;
        if (iy >= 0 && iy < 27 && ix >= 0 && ix < 27)
          v = smp[(size_t)c * 729 + iy * 27 + ix];
      }
      d[i] = f2bf(v);
    }
  }
};
struct AConv2 {            // im2col of s_pool bf16 NHWC [p][49][128]; k = (ky*3+kx)*128 + c
  const unsigned short* sp; const unsigned short* base; int oy, ox;
  __device__ __forceinline__ void prep(int row) {
    int p = row / 49, r = row % 49;
    base = sp + (size_t)p * 49 * 128;
    oy = r / 7; ox = r % 7;
  }
  __device__ __forceinline__ void load8(int k, unsigned short* d) const {
    int t = k >> 7, c = k & 127;                 // 8-run never crosses a 128-c block
    int ky = t / 3, kx = t % 3;
    int iy = oy - 1 + ky, ix = ox - 1 + kx;
    if (iy >= 0 && iy < 7 && ix >= 0 && ix < 7) {
      *(u32x4*)d = *(const u32x4*)(base + (iy * 7 + ix) * 128 + c);
    } else {
      u32x4 z = {0u, 0u, 0u, 0u}; *(u32x4*)d = z;
    }
  }
};

// ---------------- Epilogues ----------------
struct EpiF32 {            // C f32, +bias, strided concat store
  float* C; const float* bias; int ldc; int colOff;
  __device__ __forceinline__ void store(int row, int col, float v) const {
    C[(size_t)row * ldc + colOff + col] = v + bias[col];
  }
};
struct EpiBF16Relu {       // relu(v + bias) -> bf16
  unsigned short* C; const float* bias; int ldc;
  __device__ __forceinline__ void store(int row, int col, float v) const {
    float x = v + bias[col]; if (x < 0.f) x = 0.f;
    C[(size_t)row * ldc + col] = f2bf(x);
  }
};
struct EpiUnion {          // us = (u + b_union) + BN2(z)  -> bf16
  unsigned short* C; const float* bias; const unsigned short* z;
  const float* sc; const float* sh;
  __device__ __forceinline__ void store(int row, int col, float v) const {
    size_t i = (size_t)row * 256 + col;
    float s = bf2f(z[i]) * sc[col] + sh[col];
    C[i] = f2bf(v + bias[col] + s);
  }
};

// ---------------- WMMA GEMM core ----------------
// Block = 4 waves = 64 rows x (CT*16) cols. B tile (CT*16 x 32k bf16) is staged into
// LDS with async global->LDS copies, double-buffered so stage i+1 fill overlaps the
// WMMAs of stage i. Per step: batch all ds_load_b128 B-fragments first (single DScnt
// wait), then run the back-to-back WMMA chain with the reuse-A operand hint.
template <int CT, class AF, class EF>
__global__ void __launch_bounds__(128) gemm_bf16_lds(AF af, EF ef, const unsigned short* Bt,
                                                     int M, int N, int K) {
  __shared__ __attribute__((aligned(16))) unsigned short bt[2][CT * 16 * 32];
  const int tid = threadIdx.x;
  const int lane = tid & 31, wave = tid >> 5, l16 = lane & 15, lh = lane >> 4;
  const int rowTile = blockIdx.x * 64 + wave * 16;
  const int colBase = blockIdx.y * (CT * 16);
  int arow = rowTile + l16; if (arow >= M) arow = M - 1;   // stores are row-guarded
  af.prep(arow);

  const int nsteps = K / 32;
  auto stage = [&](int kk, int buf) {
    unsigned ldsBase = (unsigned)(uintptr_t)(&bt[buf][0]);
#pragma unroll
    for (int c = 0; c < CT / 2; ++c) {             // CT*64 16B-chunks, 128 threads
      int ch = c * 128 + tid;
      int nl = ch >> 2, q = ch & 3;                // tile row nl (32 bf16 = 4 chunks)
      const unsigned short* g = Bt + (size_t)(colBase + nl) * K + kk + q * 8;
      async_ld_b128(ldsBase + (unsigned)(nl * 64 + q * 16), g);
    }
  };
  stage(0, 0);

  v8f acc[CT] = {};
  for (int i = 0; i < nsteps; ++i) {
    wait_async0();                                 // my stage-i chunks landed
    __syncthreads();                               // everyone's landed; prev compute done
    if (i + 1 < nsteps) stage((i + 1) * 32, (i + 1) & 1);  // overlaps with compute below
    const int kk = i * 32;
    Frag a;
    af.load8(kk + 8 * lh, a.us);                   // ISA 16-bit A layout: 2 runs of 8
    af.load8(kk + 8 * lh + 16, a.us + 8);
    const unsigned short* bbase = &bt[i & 1][0] + l16 * 32 + lh * 8;
    Frag bf[CT];                                   // batch all B-frag ds_loads first
#pragma unroll
    for (int j = 0; j < CT; ++j) {
      const unsigned short* bp = bbase + j * 512;
      bf[j].q[0] = *(const u32x4*)(bp);
      bf[j].q[1] = *(const u32x4*)(bp + 16);
    }
    // back-to-back WMMA chain; instructions 1..CT-1 reuse the cached A operand
    acc[0] = __builtin_amdgcn_wmma_f32_16x16x32_bf16(false, a.v, false, bf[0].v,
                                                     (short)0, acc[0], false, false);
#pragma unroll
    for (int j = 1; j < CT; ++j)
      acc[j] = __builtin_amdgcn_wmma_f32_16x16x32_bf16(false, a.v, false, bf[j].v,
                                                       (short)0, acc[j], true, false);
  }
#pragma unroll
  for (int j = 0; j < CT; ++j) {
#pragma unroll
    for (int v = 0; v < 8; ++v) {
      int row = rowTile + v + 8 * lh;              // C layout: lanes 0-15 M=v, 16-31 M=v+8
      int col = colBase + j * 16 + l16;
      if (row < M) ef.store(row, col, acc[j][v]);
    }
  }
}

// ---------------- Conversion / packing kernels ----------------
__global__ void cvt_bf16_k(const float* in, unsigned short* out, int n) {
  int i = blockIdx.x * blockDim.x + threadIdx.x;
  if (i < n) out[i] = f2bf(in[i]);
}
__global__ void pack_tr_k(const float* W, unsigned short* Bt, int K, int N) { // Bt[n][k]=W[k][n]
  int i = blockIdx.x * blockDim.x + threadIdx.x;
  if (i >= K * N) return;
  int n = i / K, k = i % K;
  Bt[i] = f2bf(W[(size_t)k * N + n]);
}
__global__ void pack_c1_k(const float* w, unsigned short* Bt) {               // [128][128], k=c*49+t
  int i = blockIdx.x * blockDim.x + threadIdx.x;
  if (i >= 128 * 128) return;
  int o = i / 128, k = i % 128;
  float v = 0.f;
  if (k < 98) { int c = k / 49, t = k % 49; v = w[((size_t)o * 2 + c) * 49 + t]; }
  Bt[i] = f2bf(v);
}
__global__ void pack_c2_k(const float* w, unsigned short* Bt) {               // [256][1152], k=t*128+c
  int i = blockIdx.x * blockDim.x + threadIdx.x;
  if (i >= 256 * 1152) return;
  int o = i / 1152, k = i % 1152, t = k >> 7, c = k & 127;
  Bt[i] = f2bf(w[((size_t)o * 128 + c) * 9 + t]);
}
__global__ void pack_vr_k(const float* w, unsigned short* Bt) {               // [512][12544], k=hw*256+c
  int i = blockIdx.x * blockDim.x + threadIdx.x;
  if (i >= 512 * 12544) return;
  int n = i / 12544, k = i % 12544, hw = k >> 8, c = k & 255;
  Bt[i] = f2bf(w[((size_t)c * 49 + hw) * 512 + n]);
}

// ---------------- BatchNorm (training-mode batch stats, deterministic 2-stage) ----------------
__global__ void bnstats_k(const unsigned short* x, int M, int C, float* psum, float* psq) {
  int c = threadIdx.x;                 // blockDim == C
  int r0 = blockIdx.x * 512;
  float s = 0.f, q = 0.f;
  for (int i = 0; i < 512; ++i) {
    int r = r0 + i;
    if (r < M) { float v = bf2f(x[(size_t)r * C + c]); s += v; q += v * v; }
  }
  psum[(size_t)blockIdx.x * C + c] = s;
  psq [(size_t)blockIdx.x * C + c] = q;
}
__global__ void bnfin_k(const float* psum, const float* psq, int nblk, int C, float invM,
                        const float* g, const float* be, float* sc, float* sh) {
  int c = blockIdx.x * blockDim.x + threadIdx.x;
  if (c >= C) return;
  float s = 0.f, q = 0.f;
  for (int i = 0; i < nblk; ++i) { s += psum[(size_t)i * C + c]; q += psq[(size_t)i * C + c]; }
  float m = s * invM, v = q * invM - m * m;
  float scale = g[c] * rsqrtf(v + EPSV);
  sc[c] = scale; sh[c] = be[c] - m * scale;
}

// maxpool 3x3 s2 p1 of BN1(x1), NHWC 14x14x128 -> 7x7x128 bf16
__global__ void maxpool_k(const unsigned short* x1, const float* sc, const float* sh,
                          unsigned short* sp) {
  int i = blockIdx.x * blockDim.x + threadIdx.x;
  if (i >= N_PAIR * 49 * 128) return;
  int c = i & 127, r = i >> 7, ohw = r % 49, p = r / 49, oy = ohw / 7, ox = ohw % 7;
  float scv = sc[c], shv = sh[c], mx = -3.4e38f;
  for (int ky = 0; ky < 3; ++ky) {
    int iy = oy * 2 - 1 + ky; if (iy < 0 || iy >= 14) continue;
    for (int kx = 0; kx < 3; ++kx) {
      int ix = ox * 2 - 1 + kx; if (ix < 0 || ix >= 14) continue;
      float v = bf2f(x1[((size_t)p * 196 + iy * 14 + ix) * 128 + c]) * scv + shv;
      if (v > mx) mx = v;
    }
  }
  sp[i] = f2bf(mx);
}

// ragged mean over images (im_idx = repeat(arange(100),50): 50 rows/image)
__global__ void segmean_k(const float* xv, float* pooled) {
  int b = blockIdx.x;
  for (int j = threadIdx.x; j < 1536; j += blockDim.x) {
    float s = 0.f;
    for (int i = 0; i < PPI; ++i) s += xv[((size_t)b * PPI + i) * 1536 + j];
    pooled[(size_t)b * 1536 + j] = s * (1.f / PPI);
  }
}
__global__ void head1_k(const float* pooled, const float* w, const float* b, float* h) {
  int i = blockIdx.x * blockDim.x + threadIdx.x;
  if (i >= N_IMG * 2048) return;
  int bi = i / 2048, n = i % 2048;
  float s = b[n];
  const float* pr = pooled + (size_t)bi * 1536;
  for (int k = 0; k < 1536; ++k) s += pr[k] * w[(size_t)k * 2048 + n];
  h[i] = s > 0.f ? s : 0.f;
}
__global__ void head2_k(const float* h, const float* w, const float* b, float* out) {
  int i = blockIdx.x * blockDim.x + threadIdx.x;
  if (i >= N_IMG * 157) return;
  int bi = i / 157, n = i % 157;
  float s = b[n];
  const float* hr = h + (size_t)bi * 2048;
  for (int k = 0; k < 2048; ++k) s += hr[k] * w[(size_t)k * 157 + n];
  out[i] = 1.f / (1.f + expf(-s));
}

// ---------------- Host launch ----------------
static size_t alignUp(size_t x) { return (x + 255) & ~(size_t)255; }

extern "C" void kernel_launch(void* const* d_in, const int* in_sizes, int n_in,
                              void* d_out, int out_size, void* d_ws, size_t ws_size,
                              hipStream_t stream) {
  (void)in_sizes; (void)n_in; (void)out_size; (void)ws_size;
  const float* features = (const float*)d_in[0];
  const float* union_feat = (const float*)d_in[1];
  const float* spatial = (const float*)d_in[2];
  const float* w_union = (const float*)d_in[3];
  const float* b_union = (const float*)d_in[4];
  const float* w_c1 = (const float*)d_in[5];
  const float* b_c1 = (const float*)d_in[6];
  const float* g_bn1 = (const float*)d_in[7];
  const float* be_bn1 = (const float*)d_in[8];
  const float* w_c2 = (const float*)d_in[9];
  const float* b_c2 = (const float*)d_in[10];
  const float* g_bn2 = (const float*)d_in[11];
  const float* be_bn2 = (const float*)d_in[12];
  const float* w_subj = (const float*)d_in[13];
  const float* b_subj = (const float*)d_in[14];
  const float* w_obj = (const float*)d_in[15];
  const float* b_obj = (const float*)d_in[16];
  const float* w_vr = (const float*)d_in[17];
  const float* b_vr = (const float*)d_in[18];
  const float* w_m1 = (const float*)d_in[19];
  const float* b_m1 = (const float*)d_in[20];
  const float* w_m2 = (const float*)d_in[21];
  const float* b_m2 = (const float*)d_in[22];
  const int* pair_idx = (const int*)d_in[23];
  float* outp = (float*)d_out;

  // ---- workspace layout ----
  char* ws = (char*)d_ws; size_t cur = 0;
  auto take = [&](size_t bytes) { char* p = ws + cur; cur = alignUp(cur + bytes); return p; };
  const int M1 = N_PAIR * 196;   // 980000 conv1 rows
  const int M2 = N_PAIR * 49;    // 245000 conv2/union rows
  const int nblk1 = (M1 + 511) / 512;
  const int nblk2 = (M2 + 511) / 512;

  unsigned short* featB = (unsigned short*)take((size_t)6000 * 2048 * 2);
  unsigned short* BtS   = (unsigned short*)take((size_t)512 * 2048 * 2);
  unsigned short* BtO   = (unsigned short*)take((size_t)512 * 2048 * 2);
  unsigned short* BtU   = (unsigned short*)take((size_t)256 * 1024 * 2);
  unsigned short* BtC1  = (unsigned short*)take((size_t)128 * 128 * 2);
  unsigned short* BtC2  = (unsigned short*)take((size_t)256 * 1152 * 2);
  unsigned short* BtVR  = (unsigned short*)take((size_t)512 * 12544 * 2);
  unsigned short* spool = (unsigned short*)take((size_t)M2 * 128 * 2);
  float* xv     = (float*)take((size_t)N_PAIR * 1536 * 4);
  float* pooled = (float*)take((size_t)N_IMG * 1536 * 4);
  float* hbuf   = (float*)take((size_t)N_IMG * 2048 * 4);
  float* p1sum  = (float*)take((size_t)nblk1 * 128 * 4);
  float* p1sq   = (float*)take((size_t)nblk1 * 128 * 4);
  float* p2sum  = (float*)take((size_t)nblk2 * 256 * 4);
  float* p2sq   = (float*)take((size_t)nblk2 * 256 * 4);
  float* sc1 = (float*)take(128 * 4);  float* sh1 = (float*)take(128 * 4);
  float* sc2 = (float*)take(256 * 4);  float* sh2 = (float*)take(256 * 4);
  char* big = take((size_t)M1 * 128 * 2);            // 251 MB, reused
  unsigned short* x1 = (unsigned short*)big;         // [M1][128] bf16 (phase 1)
  unsigned short* zb = (unsigned short*)big;         // [M2][256] bf16 (phase 2, x1 dead)
  unsigned short* us = (unsigned short*)(big + (size_t)M2 * 256 * 2);

  const int T = 256;
  auto G = [&](size_t n) { return (unsigned)((n + T - 1) / T); };

  // ---- weight/activation conversion ----
  cvt_bf16_k<<<G((size_t)6000 * 2048), T, 0, stream>>>(features, featB, 6000 * 2048);
  pack_tr_k<<<G((size_t)512 * 2048), T, 0, stream>>>(w_subj, BtS, 2048, 512);
  pack_tr_k<<<G((size_t)512 * 2048), T, 0, stream>>>(w_obj, BtO, 2048, 512);
  cvt_bf16_k<<<G((size_t)256 * 1024), T, 0, stream>>>(w_union, BtU, 256 * 1024); // already [N][K]
  pack_c1_k<<<G(128 * 128), T, 0, stream>>>(w_c1, BtC1);
  pack_c2_k<<<G(256 * 1152), T, 0, stream>>>(w_c2, BtC2);
  pack_vr_k<<<G((size_t)512 * 12544), T, 0, stream>>>(w_vr, BtVR);

  // ---- conv1 (im2col WMMA) -> relu -> x1 bf16 NHWC [M1][128] ----
  {
    dim3 grid((M1 + 63) / 64, 1);   // N=128 in one pass
    gemm_bf16_lds<8, AConv1, EpiBF16Relu><<<grid, 128, 0, stream>>>(
        AConv1{spatial}, EpiBF16Relu{x1, b_c1, 128}, BtC1, M1, 128, 128);
  }
  // ---- BN1 stats -> scale/shift; fuse affine into maxpool ----
  bnstats_k<<<nblk1, 128, 0, stream>>>(x1, M1, 128, p1sum, p1sq);
  bnfin_k<<<1, 128, 0, stream>>>(p1sum, p1sq, nblk1, 128, 1.f / (float)M1, g_bn1, be_bn1, sc1, sh1);
  maxpool_k<<<G((size_t)N_PAIR * 49 * 128), T, 0, stream>>>(x1, sc1, sh1, spool);

  // ---- conv2 (im2col WMMA) -> relu -> z bf16 NHWC [M2][256] (overwrites x1 region) ----
  {
    dim3 grid((M2 + 63) / 64, 2);
    gemm_bf16_lds<8, AConv2, EpiBF16Relu><<<grid, 128, 0, stream>>>(
        AConv2{spool}, EpiBF16Relu{zb, b_c2, 256}, BtC2, M2, 256, 1152);
  }
  // ---- BN2 stats ----
  bnstats_k<<<nblk2, 256, 0, stream>>>(zb, M2, 256, p2sum, p2sq);
  bnfin_k<<<1, 256, 0, stream>>>(p2sum, p2sq, nblk2, 256, 1.f / (float)M2, g_bn2, be_bn2, sc2, sh2);

  // ---- union 1x1 conv (WMMA) + fused (u + b_union + BN2(z)) -> us bf16 [M2][256] ----
  {
    dim3 grid((M2 + 63) / 64, 2);
    gemm_bf16_lds<8, AUnion, EpiUnion><<<grid, 128, 0, stream>>>(
        AUnion{union_feat}, EpiUnion{us, b_union, zb, sc2, sh2}, BtU, M2, 256, 1024);
  }
  // ---- subj / obj gathered projections -> xv[:, 0:512], [512:1024] ----
  {
    dim3 grid((N_PAIR + 63) / 64, 4);
    gemm_bf16_lds<8, AGather, EpiF32><<<grid, 128, 0, stream>>>(
        AGather{featB, pair_idx, 0}, EpiF32{xv, b_subj, 1536, 0}, BtS, N_PAIR, 512, 2048);
    gemm_bf16_lds<8, AGather, EpiF32><<<grid, 128, 0, stream>>>(
        AGather{featB, pair_idx, 1}, EpiF32{xv, b_obj, 1536, 512}, BtO, N_PAIR, 512, 2048);
    // ---- vr = us @ w_vr (repacked to NHWC ordering) -> xv[:, 1024:1536] ----
    gemm_bf16_lds<8, ADirect, EpiF32><<<grid, 128, 0, stream>>>(
        ADirect{us, 12544}, EpiF32{xv, b_vr, 1536, 1024}, BtVR, N_PAIR, 512, 12544);
  }
  // ---- segment mean + MLP head (f32 scalar; ~0.7 GFLOP) ----
  segmean_k<<<N_IMG, 256, 0, stream>>>(xv, pooled);
  head1_k<<<G((size_t)N_IMG * 2048), T, 0, stream>>>(pooled, w_m1, b_m1, hbuf);
  head2_k<<<G((size_t)N_IMG * 157), T, 0, stream>>>(hbuf, w_m2, b_m2, outp);
}
